// CorrelatedAttentionBlock_50654844289158
// MI455X (gfx1250) — compile-verified, matrix operand order checked by reading, not run on hardware
//
#include <hip/hip_runtime.h>
#include <hip/hip_bf16.h>

// ---------------------------------------------------------------------------
// CorrelatedAttentionBlock for MI455X (gfx1250, wave32, WMMA).
// bf16 storage / WMMA inputs, f32 accumulation. All WMMA fragments are
// gathered with contiguous 16B-per-lane loads (ds_load_b128 / global_load_b128).
// The output-GEMM A tiles are staged with GLOBAL_LOAD_ASYNC_TO_LDS_B128
// (ASYNCcnt) so staging bypasses the VGPR file.
// ---------------------------------------------------------------------------

typedef __bf16 bf16;
typedef __attribute__((ext_vector_type(16))) __bf16 v16bf;
typedef __attribute__((ext_vector_type(8)))  __bf16 v8bf;
typedef __attribute__((ext_vector_type(8)))  float  v8f;

constexpr int Bn  = 2;
constexpr int Cn  = 256;
constexpr int Tn  = 256;
constexpr int Fn  = 128;
constexpr int Hn  = 8;
constexpr int Dh  = 32;
constexpr float EPSv = 1e-8f;
constexpr float MAXLAG = 0.2f;

__device__ __forceinline__ v8f wmma_bf16(v16bf a, v16bf b, v8f c) {
  return __builtin_amdgcn_wmma_f32_16x16x32_bf16(
      /*neg_a=*/false, a, /*neg_b=*/false, b,
      /*c_mod=*/(short)0, c, /*reuse_a=*/false, /*reuse_b=*/false);
}

// bf16 pair pack (lowers to v_cvt_pk_bf16_f32).
__device__ __forceinline__ unsigned pk2(float a, float b) {
  union { __bf16 h[2]; unsigned u; } c;
  c.h[0] = (__bf16)a; c.h[1] = (__bf16)b;
  return c.u;
}

// Generic LDS pointer -> 32-bit DS address (ISA 10.2: LDS generic address is
// {SHARED_BASE[63:32], offset[31:0]}; hardware truncates to 32 bits).
__device__ __forceinline__ unsigned lds_addr(const void* p) {
  return (unsigned)(unsigned long long)p;
}

// Async 16B global->LDS copy (VGLOBAL op 98, GV mode), tracked by ASYNCcnt.
__device__ __forceinline__ void async_g2l_b128(unsigned lds, const void* gsrc) {
  asm volatile("global_load_async_to_lds_b128 %0, %1, off"
               :: "v"(lds), "v"((unsigned long long)gsrc)
               : "memory");
}
__device__ __forceinline__ void wait_async0() {
  asm volatile("s_wait_asynccnt 0x0" ::: "memory");
}

// 16x32 fragment (ISA 7.12.2, wave32): lane<16 -> i=lane, k in {0..7,16..23};
// lane>=16 -> i=lane-16, k in {8..15,24..31}. Element (i,k) at p[i*rs + k].
// Per lane: two contiguous 16B loads + a register shuffle.
//   A (MxK row-major, ld)          : frag_ld_contig(p, ld)
//   B as (NxK row-major, ld), i:=n : frag_ld_contig(p, ld)
__device__ __forceinline__ v16bf frag_ld_contig(const bf16* p, int rs) {
  const int lane = threadIdx.x & 31;
  const int i  = lane & 15;
  const int kh = ((lane >> 4) & 1) * 8;
  v8bf lo = *(const v8bf*)(p + i * rs + kh);        // k = kh .. kh+7
  v8bf hi = *(const v8bf*)(p + i * rs + 16 + kh);   // k = 16+kh .. 16+kh+7
  return __builtin_shufflevector(lo, hi, 0, 1, 2, 3, 4, 5, 6, 7,
                                 8, 9, 10, 11, 12, 13, 14, 15);
}

// f32 16x16 C/D layout: VGPR v -> M = v + 8*(lane>=16), N = lane&15.
__device__ __forceinline__ void st_f32(float* p, int ld, v8f c) {
  const int lane = threadIdx.x & 31;
  const int n  = lane & 15;
  const int mb = ((lane >> 4) & 1) * 8;
#pragma unroll
  for (int v = 0; v < 8; ++v) p[(mb + v) * ld + n] = c[v];
}
__device__ __forceinline__ void st_bf16(bf16* p, int ld, v8f c) {
  const int lane = threadIdx.x & 31;
  const int n  = lane & 15;
  const int mb = ((lane >> 4) & 1) * 8;
#pragma unroll
  for (int v = 0; v < 8; ++v) p[(mb + v) * ld + n] = (bf16)c[v];
}

// Pack 32 floats -> 32 bf16 and store as 4 x uint4 (ds_store_b128 x4).
__device__ __forceinline__ void st_row32(bf16* dst, const float* v) {
#pragma unroll
  for (int i = 0; i < 4; ++i) {
    uint4 u;
    u.x = pk2(v[8*i + 0], v[8*i + 1]);
    u.y = pk2(v[8*i + 2], v[8*i + 3]);
    u.z = pk2(v[8*i + 4], v[8*i + 5]);
    u.w = pk2(v[8*i + 6], v[8*i + 7]);
    *((uint4*)(dst + 8 * i)) = u;
  }
}

// ---------------------------------------------------------------------------
// Kernel 1: projection GEMM  Y[r,c'] = sum_c x[b,c,t,f] * W[c',c] + bias[c']
// dmajor=1 (Q,K): write (B,H,F,Dh,T);  dmajor=0 (V): write (B,H,F,T,Dh).
// Grid (65536/64, 256/128), 256 threads (8 waves, 2x4, 32x32 per wave).
// ---------------------------------------------------------------------------
__global__ __launch_bounds__(256)
void proj_gemm_kernel(const float* __restrict__ x, const float* __restrict__ w,
                      const float* __restrict__ bias, bf16* __restrict__ out,
                      int dmajor) {
  __shared__ bf16 aS[64 * 32];    // [m=f][k=c]
  __shared__ bf16 bS[128 * 32];   // [n][k]  (k-major: contiguous frags)

  const int tid = threadIdx.x;
  const int r0  = blockIdx.x * 64;
  const int n0g = blockIdx.y * 128;
  const int b   = r0 >> 15;
  const int t   = (r0 >> 7) & (Tn - 1);
  const int f0  = r0 & (Fn - 1);            // tile stays inside one (b,t)

  const int w_id = tid >> 5;
  const int wm = w_id & 1, wn = w_id >> 1;
  const int m0 = wm * 32, n0w = wn * 32;

  v8f acc[2][2] = {};

  for (int kk = 0; kk < 8; ++kk) {
    const int c0 = kk * 32;
    { // A tile: aS[f][c] = x[b, c0+k, t, f0+f]; global-coalesced along f
      const int k  = tid >> 3;
      const int fb = (tid & 7) * 8;
      const float* src = x + (((size_t)(b * Cn + c0 + k) * Tn + t) * Fn) + f0 + fb;
      if (kk < 7) __builtin_prefetch(src + (size_t)32 * Tn * Fn, 0, 1);
#pragma unroll
      for (int j = 0; j < 8; ++j) aS[(fb + j) * 32 + k] = (bf16)src[j];
    }
    { // B tile: bS[n][k] = W[n0g+n, c0+k]; contiguous load + packed store
      const int n  = tid >> 1;
      const int kb = (tid & 1) * 16;
      const float* src = w + (size_t)(n0g + n) * Cn + c0 + kb;
      unsigned* dst = (unsigned*)(bS + n * 32 + kb);
#pragma unroll
      for (int j = 0; j < 8; ++j) dst[j] = pk2(src[2*j], src[2*j + 1]);
    }
    __syncthreads();

    v16bf a0 = frag_ld_contig(aS + (m0 + 0)  * 32, 32);
    v16bf a1 = frag_ld_contig(aS + (m0 + 16) * 32, 32);
    v16bf b0 = frag_ld_contig(bS + (n0w + 0)  * 32, 32);
    v16bf b1 = frag_ld_contig(bS + (n0w + 16) * 32, 32);
    acc[0][0] = wmma_bf16(a0, b0, acc[0][0]);
    acc[0][1] = wmma_bf16(a0, b1, acc[0][1]);
    acc[1][0] = wmma_bf16(a1, b0, acc[1][0]);
    acc[1][1] = wmma_bf16(a1, b1, acc[1][1]);
    __syncthreads();
  }

  const int lane = tid & 31;
  const int n_l  = lane & 15;
  const int mb   = ((lane >> 4) & 1) * 8;
#pragma unroll
  for (int i = 0; i < 2; ++i)
#pragma unroll
    for (int j = 0; j < 2; ++j) {
      v8f c = acc[i][j];
#pragma unroll
      for (int v = 0; v < 8; ++v) {
        const int m  = m0 + i * 16 + mb + v;
        const int cg = n0g + n0w + j * 16 + n_l;
        const float val = c[v] + bias[cg];
        const int h = cg >> 5, d = cg & 31;
        const int f = f0 + m;
        const size_t inst = (size_t)(b * Hn + h) * Fn + f;
        const size_t idx = dmajor ? (inst * Dh + d) * Tn + t
                                  : (inst * Tn + t) * Dh + d;
        out[idx] = (bf16)val;
      }
    }
}

// ---------------------------------------------------------------------------
// Kernel 2: fused per-(b,h,f) attention core (2048 blocks x 256 threads).
// Q,K arrive d-major [Dh][T]; V t-major [T][Dh]. Everything between the
// projections and the merged-head output lives in LDS/registers.
// ---------------------------------------------------------------------------
__global__ __launch_bounds__(256)
void attn_fused_kernel(const bf16* __restrict__ qw, const bf16* __restrict__ kw,
                       const bf16* __restrict__ vw,
                       const float* __restrict__ log_tau,
                       const float* __restrict__ lag_raw,
                       const float* __restrict__ gate_w,
                       const float* __restrict__ gate_b,
                       bf16* __restrict__ ows) {
  // 48KB region aliased across phases:
  //   phase A: qT/kT/ksT  [Dh][T] bf16 (16KB each)
  //   phase B: o0/o1      [T][Dh] bf16 (16KB each)
  __shared__ __align__(16) unsigned char smem[48 * 1024];
  bf16* qT  = (bf16*)smem;
  bf16* kT  = qT + Dh * Tn;
  bf16* ksT = kT + Dh * Tn;
  bf16* o0  = (bf16*)smem;            // alias qT
  bf16* o1  = o0 + Tn * Dh;           // alias kT
  __shared__ __align__(16) float covS[2 * 32 * 32];   // later: gate_w bf16 cache
  __shared__ __align__(16) bf16  attT[2 * 32 * 32];   // [br][e][d] (k-major)
  __shared__ float red[2][32][8];
  __shared__ float rsqS[2][32];

  const int tid = threadIdx.x;
  const int bid = blockIdx.x;
  const int b = bid >> 10;
  const int h = (bid >> 7) & (Hn - 1);
  const int f = bid & (Fn - 1);
  const size_t base = (size_t)bid * Tn * Dh;
  const bf16* qg = qw + base;   // [Dh][T]
  const bf16* kg = kw + base;   // [Dh][T]
  const bf16* vg = vw + base;   // [T][Dh]

  const int d  = tid & 31;
  const int tc = tid >> 5;
  const int t0 = tc * 32;       // this thread owns (d, t0..t0+31)

  // --- load rows into registers (2 x 32B vector loads each) ----------------
  float qv[32], kv[32];
  {
    const v16bf* qp = (const v16bf*)(qg + d * Tn + t0);
    const v16bf* kp = (const v16bf*)(kg + d * Tn + t0);
    v16bf q0 = qp[0], q1 = qp[1], k0 = kp[0], k1 = kp[1];
#pragma unroll
    for (int i = 0; i < 16; ++i) {
      qv[i] = (float)q0[i]; qv[16 + i] = (float)q1[i];
      kv[i] = (float)k0[i]; kv[16 + i] = (float)k1[i];
    }
  }
  float sq = 0.f, sk = 0.f;
#pragma unroll
  for (int i = 0; i < 32; ++i) { sq += qv[i] * qv[i]; sk += kv[i] * kv[i]; }
  red[0][d][tc] = sq;
  red[1][d][tc] = sk;
  __syncthreads();
  if (tid < 64) {
    const int which = tid >> 5, dd = tid & 31;
    float s = 0.f;
#pragma unroll
    for (int j = 0; j < 8; ++j) s += red[which][dd][j];
    rsqS[which][dd] = rsqrtf(fmaxf(s, EPSv));       // 1/sqrt(clip(sum,EPS))
  }
  __syncthreads();

  // --- normalize in registers, packed store to LDS -------------------------
  const float rq = rsqS[0][d], rk = rsqS[1][d];
#pragma unroll
  for (int i = 0; i < 32; ++i) { qv[i] *= rq; kv[i] *= rk; }
  st_row32(qT + d * Tn + t0, qv);
  st_row32(kT + d * Tn + t0, kv);
  __syncthreads();

  // --- fractional lag shift (lag in (-0.2,0.2) => only t+-1 neighbors) -----
  const float lag = MAXLAG * tanhf(lag_raw[h * Dh + d]);
  const float prev = (t0 == 0)        ? kv[0]  : (float)kT[d * Tn + t0 - 1];
  const float nxt  = (t0 + 32 >= Tn)  ? kv[31] : (float)kT[d * Tn + t0 + 32];
  float ksv[32];
#pragma unroll
  for (int i = 0; i < 32; ++i) {
    const int t = t0 + i;
    const float km1 = i ? kv[i - 1] : prev;
    const float kp1 = (i < 31) ? kv[i + 1] : nxt;
    float v;
    if (lag > 0.f) v = (t == 0)      ? kv[0] : (lag * km1 + (1.f - lag) * kv[i]);
    else           v = (t == Tn - 1) ? kv[i] : ((1.f + lag) * kv[i] - lag * kp1);
    ksv[i] = v;
  }
  st_row32(ksT + d * Tn + t0, ksv);
  __syncthreads();

  // --- cov GEMMs: cov[br][d][e] = sum_t Ksrc[d,t] * Qhat[e,t] --------------
  {
    const int w  = tid >> 5;
    const int br = w >> 2;
    const int m0 = ((w >> 1) & 1) * 16;
    const int n0 = (w & 1) * 16;
    const bf16* A = (br ? ksT : kT) + m0 * Tn;
    const bf16* Bq = qT + n0 * Tn;
    v8f c = {};
#pragma unroll
    for (int kk = 0; kk < 8; ++kk) {
      v16bf av = frag_ld_contig(A  + kk * 32, Tn);
      v16bf bv = frag_ld_contig(Bq + kk * 32, Tn);
      c = wmma_bf16(av, bv, c);
    }
    st_f32(covS + br * 1024 + m0 * 32 + n0, 32, c);
  }
  __syncthreads();

  // --- softmax over e (scaled 1/tau), store att transposed [e][d] ----------
  const float tauv = fminf(fmaxf(__expf(log_tau[0]), 1e-4f), 10.f);
  const float itau = 1.f / tauv;
  if (tid < 64) {
    const int br = tid >> 5, dd = tid & 31;
    const float* row = covS + br * 1024 + dd * 32;
    float mx = -3.4e38f;
#pragma unroll
    for (int e = 0; e < 32; ++e) mx = fmaxf(mx, row[e]);
    float p[32];
    float sum = 0.f;
#pragma unroll
    for (int e = 0; e < 32; ++e) { p[e] = __expf((row[e] - mx) * itau); sum += p[e]; }
    const float inv = 1.f / sum;
#pragma unroll
    for (int e = 0; e < 32; ++e)
      attT[br * 1024 + e * 32 + dd] = (bf16)(p[e] * inv);   // [e][d]
  }
  __syncthreads();

  // --- cache gate_w (bf16, [e][2Dh]) in the dead cov buffer ----------------
  bf16* gwT = (bf16*)covS;    // 32 x 64 bf16 = 4KB (covS is 8KB)
  {
    unsigned* gp = (unsigned*)gwT;
#pragma unroll
    for (int i = 0; i < 4; ++i) {
      const int u = tid + i * 256;
      gp[u] = pk2(gate_w[2 * u], gate_w[2 * u + 1]);
    }
  }

  // --- inst/lagged: O[t,e] = sum_d V[t,d] * att[e,d] (V frags from L2) -----
  {
    const int w = tid >> 5;
#pragma unroll
    for (int s = 0; s < 2; ++s) {
      const int tm = (w + s * 8) * 16;
      v16bf av = frag_ld_contig(vg + tm * Dh, Dh);      // global gather
#pragma unroll
      for (int br = 0; br < 2; ++br) {
        bf16* od = br ? o1 : o0;
#pragma unroll
        for (int nt = 0; nt < 2; ++nt) {
          v16bf bv = frag_ld_contig(attT + br * 1024 + nt * 16 * 32, 32);
          v8f c = {};
          c = wmma_bf16(av, bv, c);
          st_bf16(od + tm * Dh + nt * 16, Dh, c);
        }
      }
    }
  }
  __syncthreads();

  // --- gate as WMMA: logits = [o0|o1](Tx64) x gate_w^T(64x32), then blend --
  {
    const int w = tid >> 5;
    const int lane = tid & 31;
    const int n_l = lane & 15;
    const int mb  = ((lane >> 4) & 1) * 8;
#pragma unroll
    for (int s = 0; s < 2; ++s) {
      const int tm = (w + s * 8) * 16;
      v16bf alo = frag_ld_contig(o0 + tm * Dh, Dh);     // k = j in [0,32)
      v16bf ahi = frag_ld_contig(o1 + tm * Dh, Dh);     // k = j-32
#pragma unroll
      for (int nt = 0; nt < 2; ++nt) {
        v16bf blo = frag_ld_contig(gwT + nt * 16 * 64, 64);
        v16bf bhi = frag_ld_contig(gwT + nt * 16 * 64 + 32, 64);
        v8f c = {};
        c = wmma_bf16(alo, blo, c);
        c = wmma_bf16(ahi, bhi, c);
        const int e  = nt * 16 + n_l;
        const float gb = gate_b[e];
#pragma unroll
        for (int v = 0; v < 8; ++v) {
          const int t = tm + mb + v;
          const float beta = 1.f / (1.f + __expf(-(c[v] + gb)));
          const float vi = (float)o0[t * Dh + e];
          const float vl = (float)o1[t * Dh + e];
          const float ov = (1.f - beta) * vi + beta * vl;
          ows[((size_t)(b * Tn + t) * Fn + f) * Cn + h * Dh + e] = (bf16)ov;
        }
      }
    }
  }
}

// ---------------------------------------------------------------------------
// Kernel 3: output GEMM  out[b,c',t,f] = sum_c A[r,c] * wo[c',c] + bo[c']
// A bf16 (B,T,F,C): staged with GLOBAL_LOAD_ASYNC_TO_LDS_B128 (ASYNCcnt path,
// bypasses VGPRs). bS k-major.
// ---------------------------------------------------------------------------
__global__ __launch_bounds__(256)
void out_gemm_kernel(const bf16* __restrict__ aws, const float* __restrict__ w,
                     const float* __restrict__ bias, float* __restrict__ dout) {
  __shared__ bf16 aS[64 * 32];    // [m][k]
  __shared__ bf16 bS[128 * 32];   // [n][k]

  const int tid = threadIdx.x;
  const int r0  = blockIdx.x * 64;
  const int n0g = blockIdx.y * 128;
  const int b   = r0 >> 15;
  const int t   = (r0 >> 7) & (Tn - 1);
  const int f0  = r0 & (Fn - 1);

  const int w_id = tid >> 5;
  const int wm = w_id & 1, wn = w_id >> 1;
  const int m0 = wm * 32, n0w = wn * 32;

  // Per-thread staging addresses (A: 16B per thread per k-step).
  const int am  = tid >> 2;
  const int akb = (tid & 3) * 8;
  const unsigned aLds = lds_addr(aS + am * 32 + akb);
  const bf16* aSrc = aws + (size_t)(r0 + am) * Cn + akb;

  v8f acc[2][2] = {};

  for (int kk = 0; kk < 8; ++kk) {
    const int c0 = kk * 32;
    // A: async global->LDS (no VGPR round trip), tracked by ASYNCcnt
    async_g2l_b128(aLds, aSrc + c0);
    { // B: bS[n][k] = wo[n0g+n, c0+k]
      const int n  = tid >> 1;
      const int kb = (tid & 1) * 16;
      const float* src = w + (size_t)(n0g + n) * Cn + c0 + kb;
      unsigned* dst = (unsigned*)(bS + n * 32 + kb);
#pragma unroll
      for (int j = 0; j < 8; ++j) dst[j] = pk2(src[2*j], src[2*j + 1]);
    }
    wait_async0();               // s_wait_asynccnt 0
    __syncthreads();

    v16bf a0 = frag_ld_contig(aS + (m0 + 0)  * 32, 32);
    v16bf a1 = frag_ld_contig(aS + (m0 + 16) * 32, 32);
    v16bf b0 = frag_ld_contig(bS + (n0w + 0)  * 32, 32);
    v16bf b1 = frag_ld_contig(bS + (n0w + 16) * 32, 32);
    acc[0][0] = wmma_bf16(a0, b0, acc[0][0]);
    acc[0][1] = wmma_bf16(a0, b1, acc[0][1]);
    acc[1][0] = wmma_bf16(a1, b0, acc[1][0]);
    acc[1][1] = wmma_bf16(a1, b1, acc[1][1]);
    __syncthreads();
  }

  const int lane = tid & 31;
  const int n_l  = lane & 15;
  const int mb   = ((lane >> 4) & 1) * 8;
#pragma unroll
  for (int i = 0; i < 2; ++i)
#pragma unroll
    for (int j = 0; j < 2; ++j) {
      v8f c = acc[i][j];
#pragma unroll
      for (int v = 0; v < 8; ++v) {
        const int m  = m0 + i * 16 + mb + v;
        const int cg = n0g + n0w + j * 16 + n_l;
        const int f  = f0 + m;
        dout[(((size_t)(b * Cn + cg) * Tn + t) * Fn) + f] = c[v] + bias[cg];
      }
    }
}

// ---------------------------------------------------------------------------
// Host launcher
// ---------------------------------------------------------------------------
extern "C" void kernel_launch(void* const* d_in, const int* in_sizes, int n_in,
                              void* d_out, int out_size, void* d_ws, size_t ws_size,
                              hipStream_t stream) {
  (void)in_sizes; (void)n_in; (void)out_size; (void)ws_size;

  const float* x       = (const float*)d_in[0];
  const float* wq      = (const float*)d_in[1];
  const float* bq      = (const float*)d_in[2];
  const float* wk      = (const float*)d_in[3];
  const float* bk      = (const float*)d_in[4];
  const float* wv      = (const float*)d_in[5];
  const float* bv      = (const float*)d_in[6];
  const float* wo      = (const float*)d_in[7];
  const float* bo      = (const float*)d_in[8];
  const float* log_tau = (const float*)d_in[9];
  const float* lag_raw = (const float*)d_in[10];
  const float* gate_w  = (const float*)d_in[11];
  const float* gate_b  = (const float*)d_in[12];
  float* out = (float*)d_out;

  const size_t NQ = (size_t)Bn * Hn * Fn * Tn * Dh;   // 16,777,216
  bf16* qws = (bf16*)d_ws;          // (B,H,F,Dh,T)
  bf16* kws = qws + NQ;             // (B,H,F,Dh,T)
  bf16* vws = kws + NQ;             // (B,H,F,T,Dh)
  bf16* ows = vws + NQ;             // (B,T,F,C)

  const dim3 gGemm((Bn * Tn * Fn) / 64, Cn / 128);
  const dim3 blk(256);

  proj_gemm_kernel<<<gGemm, blk, 0, stream>>>(x, wq, bq, qws, 1);
  proj_gemm_kernel<<<gGemm, blk, 0, stream>>>(x, wk, bk, kws, 1);
  proj_gemm_kernel<<<gGemm, blk, 0, stream>>>(x, wv, bv, vws, 0);

  attn_fused_kernel<<<Bn * Hn * Fn, blk, 0, stream>>>(
      qws, kws, vws, log_tau, lag_raw, gate_w, gate_b, ows);

  out_gemm_kernel<<<gGemm, blk, 0, stream>>>(ows, wo, bo, out);
}
